// CombineGraph_27762668601398
// MI455X (gfx1250) — compile-verified, matrix operand order checked by reading, not run on hardware
//
#include <hip/hip_runtime.h>
#include <hip/hip_bf16.h>
#include <math.h>

#define Bq 512
#define Lq 50
#define Hq 128
#define NITEMS 249999   // emb_w[1:]

typedef __attribute__((ext_vector_type(16))) _Float16 v16h;
typedef __attribute__((ext_vector_type(8)))  _Float16 v8h;
typedef __attribute__((ext_vector_type(8)))  float    v8f;

// ---------------------------------------------------------------------------
// WMMA helpers (CDNA5 wave32, V_WMMA_F32_16X16X32_F16)
// A/B fragment for a row-major f16 tile: lane r=lane&15 selects the row
// (M for A, N for B), hi=lane>>4 selects the K-half. Elements 0..7 hold
// K = k0+8*hi .. +7, elements 8..15 hold K = k0+16+8*hi .. +7 (ISA 7.12.2).
// ---------------------------------------------------------------------------
__device__ __forceinline__ v16h load_frag(const _Float16* base, int ld,
                                          int row0, int k0, int lane) {
  const int r = lane & 15, hi = lane >> 4;
  const _Float16* p = base + (size_t)(row0 + r) * ld + k0 + hi * 8;
  v8h c0 = *(const v8h*)(p);
  v8h c1 = *(const v8h*)(p + 16);
  v16h f;
#pragma unroll
  for (int i = 0; i < 8; ++i) { f[i] = c0[i]; f[8 + i] = c1[i]; }
  return f;
}

// Same fragment but sourced from an f32 tile (convert while building).
__device__ __forceinline__ v16h load_frag_f32(const float* base, int ld,
                                              int row0, int k0, int lane) {
  const int r = lane & 15, hi = lane >> 4;
  const float* p = base + (size_t)(row0 + r) * ld + k0 + hi * 8;
  v16h f;
#pragma unroll
  for (int i = 0; i < 8; ++i) {
    f[i]     = (_Float16)p[i];
    f[8 + i] = (_Float16)p[16 + i];
  }
  return f;
}

__device__ __forceinline__ v8f wmma16(v16h a, v16h b, v8f c) {
  return __builtin_amdgcn_wmma_f32_16x16x32_f16(false, a, false, b,
                                                (short)0, c, false, false);
}

__device__ __forceinline__ float sigmoidf_(float x) {
  return 1.f / (1.f + expf(-x));
}

// Async global->LDS 16B copy (CDNA5, tracked by ASYNCcnt).
__device__ __forceinline__ void async_copy_b128(void* lds_ptr, const void* gptr) {
  unsigned lds = (unsigned)(size_t)lds_ptr;   // low 32 bits = LDS byte address
  asm volatile("global_load_async_to_lds_b128 %0, %1, off"
               :: "v"(lds), "v"(gptr) : "memory");
}
__device__ __forceinline__ void async_wait0() {
  asm volatile("s_wait_asynccnt 0x0" ::: "memory");
}

// ---------------------------------------------------------------------------
// Item inverse norms: invn[i] = 1/(||emb_w[i+1]|| + 1e-12)
// Also warms the 128MB embedding table into the 192MB L2 (RT loads).
// ---------------------------------------------------------------------------
__global__ __launch_bounds__(256) void invnorm_kernel(const float* __restrict__ emb,
                                                      float* __restrict__ invn) {
  const int w = threadIdx.x >> 5, lane = threadIdx.x & 31;
  const int item = blockIdx.x * 8 + w;
  if (item >= NITEMS) return;
  const float* row = &emb[(size_t)(item + 1) * Hq];
  float4 v = *(const float4*)&row[lane * 4];
  float s = v.x * v.x + v.y * v.y + v.z * v.z + v.w * v.w;
#pragma unroll
  for (int o = 16; o > 0; o >>= 1) s += __shfl_xor(s, o, 32);
  if (lane == 0) invn[item] = 1.f / (sqrtf(s) + 1e-12f);
}

// ---------------------------------------------------------------------------
// Per-batch prep: lengths, hs (masked mean), hsg = hs@glu2_w, ht0, lasts[0..2]
// ---------------------------------------------------------------------------
__global__ __launch_bounds__(128) void prep_kernel(const float* __restrict__ hidden,
                                                   const int* __restrict__ mask,
                                                   const float* __restrict__ glu2_w,
                                                   float* __restrict__ hs,
                                                   float* __restrict__ hsg,
                                                   float* __restrict__ ht0,
                                                   float* __restrict__ lasts) {
  const int b = blockIdx.x, t = threadIdx.x;
  __shared__ float sh[Hq];
  __shared__ int slen;
  if (t == 0) {
    int L = 0;
    for (int l = 0; l < Lq; ++l) L += mask[b * Lq + l];
    slen = L;
  }
  __syncthreads();
  const int len = slen;
  float s = 0.f;
  for (int l = 0; l < Lq; ++l)
    s += hidden[((size_t)b * Lq + l) * Hq + t] * (float)mask[b * Lq + l];
  float hv = s / (float)(len > 0 ? len : 1);
  hs[b * Hq + t] = hv;
  sh[t] = hv;
  __syncthreads();
  float g = 0.f;
  for (int i = 0; i < Hq; ++i) g += sh[i] * glu2_w[i * Hq + t];
  hsg[b * Hq + t] = g;
  int i0 = len - 1; if (i0 < 0) i0 += Lq;
  ht0[b * Hq + t] = hidden[((size_t)b * Lq + i0) * Hq + t];
  for (int j = 0; j < 3; ++j) {
    int idx = len - (j + 1);
    if (idx > 1000) idx = 1000;
    if (idx < -1) idx = -1;
    if (idx < 0) idx += Lq;
    lasts[((size_t)b * 3 + j) * Hq + t] = hidden[((size_t)b * Lq + idx) * Hq + t];
  }
}

// posproj[l][h] = pos_w[l] @ w1[0:128]   (shared across batch)
__global__ __launch_bounds__(128) void posproj_kernel(const float* __restrict__ pos_w,
                                                      const float* __restrict__ w1,
                                                      float* __restrict__ pp) {
  const int l = blockIdx.x, t = threadIdx.x;
  float s = 0.f;
  for (int i = 0; i < Hq; ++i) s += pos_w[l * Hq + i] * w1[i * Hq + t];
  pp[l * Hq + t] = s;
}

__global__ void f16_convert_kernel(const float* __restrict__ in,
                                   _Float16* __restrict__ out, int n) {
  int i = blockIdx.x * 256 + threadIdx.x;
  if (i < n) out[i] = (_Float16)in[i];
}

// Wt[n][k] = (f16) W[k][n]  for a 128x128 weight
__global__ __launch_bounds__(256) void w_transpose_f16_kernel(const float* __restrict__ W,
                                                              _Float16* __restrict__ Wt) {
  int idx = blockIdx.x * 256 + threadIdx.x;   // 16384
  int k = idx >> 7, n = idx & 127;
  Wt[n * Hq + k] = (_Float16)W[idx];
}

// ---------------------------------------------------------------------------
// Generic WMMA matmul: out(rows x 128) = epilogue( A_h(rows x 128) @ W ),
// W given pre-transposed f16 (Wt[n][k]). 8 waves: wave w owns row-tile w,
// sweeps 8 col-tiles; weight staged in LDS. rows multiple of 128.
// mode 0: nh  = tanh(acc + posproj[row%50][col])            -> f16 out
// mode 1: g   = sigmoid(acc + e0[col] + e1[row/50][col])    -> f32 out
// mode 2: out = acc + e0[col]                               -> f32 out
// ---------------------------------------------------------------------------
__global__ __launch_bounds__(256) void mm128_kernel(const _Float16* __restrict__ A,
                                                    const _Float16* __restrict__ Wt,
                                                    int mode,
                                                    const float* __restrict__ e0,
                                                    const float* __restrict__ e1,
                                                    _Float16* __restrict__ outh,
                                                    float* __restrict__ outf) {
  __shared__ __align__(16) _Float16 Ws[Hq * Hq];
  const int t = threadIdx.x;
#pragma unroll
  for (int i = 0; i < 8; ++i) {
    int idx = (t + i * 256) * 8;
    *(v8h*)&Ws[idx] = *(const v8h*)&Wt[idx];
  }
  __syncthreads();
  const int lane = t & 31, w = t >> 5;
  const int m0 = blockIdx.x * 128 + w * 16;
  const int r15 = lane & 15, hi = lane >> 4;
  v16h a0 = load_frag(A, Hq, m0, 0, lane);
  v16h a1 = load_frag(A, Hq, m0, 32, lane);
  v16h a2 = load_frag(A, Hq, m0, 64, lane);
  v16h a3 = load_frag(A, Hq, m0, 96, lane);
  for (int ct = 0; ct < 8; ++ct) {
    v8f acc = {};
    acc = wmma16(a0, load_frag(Ws, Hq, ct * 16, 0, lane), acc);
    acc = wmma16(a1, load_frag(Ws, Hq, ct * 16, 32, lane), acc);
    acc = wmma16(a2, load_frag(Ws, Hq, ct * 16, 64, lane), acc);
    acc = wmma16(a3, load_frag(Ws, Hq, ct * 16, 96, lane), acc);
    const int col = ct * 16 + r15;
#pragma unroll
    for (int r = 0; r < 8; ++r) {
      const int row = m0 + r + 8 * hi;
      float v = acc[r];
      if (mode == 0) {
        v = tanhf(v + e0[(row % Lq) * Hq + col]);
        outh[(size_t)row * Hq + col] = (_Float16)v;
      } else if (mode == 1) {
        v = sigmoidf_(v + e0[col] + e1[(row / Lq) * Hq + col]);
        outf[(size_t)row * Hq + col] = v;
      } else {
        outf[(size_t)row * Hq + col] = v + e0[col];
      }
    }
  }
}

// ---------------------------------------------------------------------------
// hts[b][i] = concat(lasts[b][0..i]) @ lq_w[i] + lq_b[i]
// ---------------------------------------------------------------------------
__global__ __launch_bounds__(128) void hts_kernel(const float* __restrict__ lasts,
                                                  const float* __restrict__ w0,
                                                  const float* __restrict__ b0,
                                                  const float* __restrict__ w1,
                                                  const float* __restrict__ b1,
                                                  const float* __restrict__ w2,
                                                  const float* __restrict__ b2,
                                                  float* __restrict__ hts) {
  const int blk = blockIdx.x, b = blk / 3, i = blk % 3, t = threadIdx.x;
  const float* W = (i == 0) ? w0 : (i == 1) ? w1 : w2;
  const float* Bv = (i == 0) ? b0 : (i == 1) ? b1 : b2;
  const int Kd = (i + 1) * Hq;
  float s = Bv[t];
  for (int j = 0; j < Kd; ++j)
    s += lasts[((size_t)b * 3 + (j >> 7)) * Hq + (j & 127)] * W[(size_t)j * Hq + t];
  hts[(size_t)blk * Hq + t] = s;
}

// normalize hts over the K axis (axis=1 of (B,3,H))
__global__ __launch_bounds__(128) void htsnorm_kernel(float* __restrict__ hts) {
  const int b = blockIdx.x, t = threadIdx.x;
  float v0 = hts[((size_t)b * 3 + 0) * Hq + t];
  float v1 = hts[((size_t)b * 3 + 1) * Hq + t];
  float v2 = hts[((size_t)b * 3 + 2) * Hq + t];
  float n = sqrtf(v0 * v0 + v1 * v1 + v2 * v2) + 1e-12f;
  hts[((size_t)b * 3 + 0) * Hq + t] = v0 / n;
  hts[((size_t)b * 3 + 1) * Hq + t] = v1 / n;
  hts[((size_t)b * 3 + 2) * Hq + t] = v2 / n;
}

// proj_z = hts @ wz + bz   (1536 rows)
__global__ __launch_bounds__(128) void projz_kernel(const float* __restrict__ hts,
                                                    const float* __restrict__ wz,
                                                    const float* __restrict__ bz,
                                                    float* __restrict__ pz) {
  const int row = blockIdx.x, t = threadIdx.x;
  float s = bz[t];
  for (int i = 0; i < Hq; ++i) s += hts[(size_t)row * Hq + i] * wz[i * Hq + t];
  pz[(size_t)row * Hq + t] = s;
}

// beta = (g @ w2) * maskf ; select = sum_l beta * hidden
__global__ __launch_bounds__(128) void beta_select_kernel(const float* __restrict__ g,
                                                          const float* __restrict__ w2,
                                                          const int* __restrict__ mask,
                                                          const float* __restrict__ hidden,
                                                          float* __restrict__ sel) {
  const int b = blockIdx.x, t = threadIdx.x;
  __shared__ float sb[Lq];
  if (t < Lq) {
    float s = 0.f;
    const float* gr = &g[((size_t)b * Lq + t) * Hq];
    for (int h = 0; h < Hq; ++h) s += gr[h] * w2[h];
    sb[t] = s * (float)mask[b * Lq + t];
  }
  __syncthreads();
  float s = 0.f;
  for (int l = 0; l < Lq; ++l) s += sb[l] * hidden[((size_t)b * Lq + l) * Hq + t];
  sel[b * Hq + t] = s;
}

// ---------------------------------------------------------------------------
// Attention / Lp-pool / second softmax / ais  (per batch row)
// ---------------------------------------------------------------------------
__global__ __launch_bounds__(256) void alpha_kernel(const float* __restrict__ pzall,
                                                    const float* __restrict__ projo,
                                                    const float* __restrict__ projt,
                                                    const int* __restrict__ mask,
                                                    float* __restrict__ ais) {
  const int b = blockIdx.x, t = threadIdx.x;
  __shared__ float pz[3 * Hq];
  __shared__ float A1[Lq * 24];
  __shared__ float A2[Lq * 8];
  for (int i = t; i < 3 * Hq; i += 256) pz[i] = pzall[(size_t)b * 3 * Hq + i];
  __syncthreads();
  // logits: sigmoid(q0 . q1), q0/q1 are flat reinterpretations
  for (int job = t; job < 24 * Lq; job += 256) {
    const int m = job / Lq, l = job % Lq;
    const int u = m / 3, k = m % 3;
    const float* q0p = &pz[u * 48 + k * 16];
    const float* q1p = &projo[(size_t)b * (Lq * Hq) + u * 800 + l * 16];
    float s = 0.f;
#pragma unroll
    for (int d = 0; d < 16; ++d) s += q0p[d] * q1p[d];
    A1[l * 24 + m] = sigmoidf_(s);
  }
  __syncthreads();
  if (t < 24) {  // softmax over L of 2*A1, per column
    float mx = -1e30f;
    for (int l = 0; l < Lq; ++l) mx = fmaxf(mx, 2.f * A1[l * 24 + t]);
    float s = 0.f;
    for (int l = 0; l < Lq; ++l) {
      float e = expf(2.f * A1[l * 24 + t] - mx);
      A1[l * 24 + t] = e; s += e;
    }
    float inv = 1.f / s;
    for (int l = 0; l < Lq; ++l) A1[l * 24 + t] *= inv;
  }
  __syncthreads();
  if (t < 8) {  // Lp pool (p=2) over K, mask, softmax over L
    float mx = -1e30f;
    for (int l = 0; l < Lq; ++l) {
      float a = A1[l * 24 + t * 3 + 0];
      float c = A1[l * 24 + t * 3 + 1];
      float d = A1[l * 24 + t * 3 + 2];
      float p = sqrtf(a * a + c * c + d * d);
      A2[l * 8 + t] = p;
      if (mask[b * Lq + l] > 0) mx = fmaxf(mx, 2.f * p);
    }
    float s = 0.f;
    for (int l = 0; l < Lq; ++l) {
      float e = (mask[b * Lq + l] > 0) ? expf(2.f * A2[l * 8 + t] - mx) : 0.f;
      A2[l * 8 + t] = e; s += e;
    }
    float inv = (s > 0.f) ? 1.f / s : 0.f;
    for (int l = 0; l < Lq; ++l) A2[l * 8 + t] *= inv;
  }
  __syncthreads();
  if (t < Hq) {  // ais[h] = sum_l A2[l][h/16] * proj_t[b,l,h] * maskf
    const int hh = t >> 4;
    float s = 0.f;
    for (int l = 0; l < Lq; ++l)
      s += A2[l * 8 + hh] * projt[((size_t)b * Lq + l) * Hq + t] * (float)mask[b * Lq + l];
    ais[b * Hq + t] = s;
  }
}

// session = normalize(0.5*(select + concat(ais,ht0)@lt2_w + lt2_b)) -> f16
__global__ __launch_bounds__(128) void session_kernel(const float* __restrict__ ais,
                                                      const float* __restrict__ ht0,
                                                      const float* __restrict__ sel,
                                                      const float* __restrict__ lt2_w,
                                                      const float* __restrict__ lt2_b,
                                                      _Float16* __restrict__ sess_h) {
  const int b = blockIdx.x, t = threadIdx.x;
  __shared__ float red[Hq];
  float a = lt2_b[t];
  for (int i = 0; i < Hq; ++i) a += ais[b * Hq + i] * lt2_w[i * Hq + t];
  for (int i = 0; i < Hq; ++i) a += ht0[b * Hq + i] * lt2_w[(Hq + i) * Hq + t];
  float ses = 0.5f * (sel[b * Hq + t] + a);
  red[t] = ses * ses;
  __syncthreads();
  for (int s = 64; s > 0; s >>= 1) {
    if (t < s) red[t] += red[t + s];
    __syncthreads();
  }
  float n = sqrtf(red[0]) + 1e-12f;
  sess_h[b * Hq + t] = (_Float16)(ses / n);
}

// ---------------------------------------------------------------------------
// Final GEMM: out(512 x 249999) = 16 * sess @ items^T * invn[n]
// Block: 128-item slab async-copied to LDS as f32 (b-frags built once with
// f32->f16 convert), 8 waves each own one 16-col tile, loop 4 session slabs
// of 128 rows async-copied to LDS (f16), 8 M-tiles each, 4 WMMA per tile.
// Output uses non-temporal stores so the 512MB write stream does not evict
// the L2-resident embedding table (128MB of 192MB L2, warmed by invnorm).
// ---------------------------------------------------------------------------
__global__ __launch_bounds__(256) void final_gemm_kernel(const float* __restrict__ emb,
                                                         const _Float16* __restrict__ sess,
                                                         const float* __restrict__ invn,
                                                         float* __restrict__ out) {
  __shared__ __align__(16) float items[128 * Hq];      // 64KB (f32 slab)
  __shared__ __align__(16) _Float16 sl[128 * Hq];      // 32KB
  const int t = threadIdx.x;
  const int nbase = blockIdx.x * 128;
  // Async-stage the item slab (clamp OOB rows; their columns are never stored)
#pragma unroll
  for (int i = 0; i < 16; ++i) {
    int j = t + i * 256;            // 16B chunk id (4096 total)
    int e = j * 4;                  // float index
    int n = nbase + (e >> 7);
    if (n >= NITEMS) n = NITEMS - 1;
    async_copy_b128(&items[e], &emb[(size_t)(n + 1) * Hq + (e & 127)]);
  }
  async_wait0();
  __syncthreads();
  const int lane = t & 31, w = t >> 5;
  const int r15 = lane & 15, hi = lane >> 4;
  v16h b0 = load_frag_f32(items, Hq, w * 16, 0, lane);
  v16h b1 = load_frag_f32(items, Hq, w * 16, 32, lane);
  v16h b2 = load_frag_f32(items, Hq, w * 16, 64, lane);
  v16h b3 = load_frag_f32(items, Hq, w * 16, 96, lane);
  const int ncol = nbase + w * 16 + r15;
  const float inv = (ncol < NITEMS) ? invn[ncol] : 0.f;
  for (int ms = 0; ms < 4; ++ms) {
    __syncthreads();
#pragma unroll
    for (int i = 0; i < 8; ++i) {
      int j = t + i * 256;          // 16B chunk id (2048 total)
      async_copy_b128(&sl[j * 8], &sess[(size_t)ms * (128 * Hq) + j * 8]);
    }
    async_wait0();
    __syncthreads();
    for (int mt = 0; mt < 8; ++mt) {
      v8f acc = {};
      acc = wmma16(load_frag(sl, Hq, mt * 16, 0, lane), b0, acc);
      acc = wmma16(load_frag(sl, Hq, mt * 16, 32, lane), b1, acc);
      acc = wmma16(load_frag(sl, Hq, mt * 16, 64, lane), b2, acc);
      acc = wmma16(load_frag(sl, Hq, mt * 16, 96, lane), b3, acc);
      if (ncol < NITEMS) {
#pragma unroll
        for (int r = 0; r < 8; ++r) {
          const int row = ms * 128 + mt * 16 + r + 8 * hi;
          __builtin_nontemporal_store(16.f * acc[r] * inv,
                                      &out[(size_t)row * NITEMS + ncol]);
        }
      }
    }
  }
}

// ---------------------------------------------------------------------------
extern "C" void kernel_launch(void* const* d_in, const int* in_sizes, int n_in,
                              void* d_out, int out_size, void* d_ws, size_t ws_size,
                              hipStream_t stream) {
  const float* hidden  = (const float*)d_in[0];
  const int*   mask    = (const int*)  d_in[1];
  const float* emb_w   = (const float*)d_in[2];
  const float* pos_w   = (const float*)d_in[3];
  const float* w1      = (const float*)d_in[4];
  const float* w2      = (const float*)d_in[5];
  const float* glu1_w  = (const float*)d_in[6];
  const float* glu1_b  = (const float*)d_in[7];
  const float* glu2_w  = (const float*)d_in[8];
  const float* lq_w0   = (const float*)d_in[9];
  const float* lq_b0   = (const float*)d_in[10];
  const float* lq_w1   = (const float*)d_in[11];
  const float* lq_b1   = (const float*)d_in[12];
  const float* lq_w2   = (const float*)d_in[13];
  const float* lq_b2   = (const float*)d_in[14];
  const float* wz      = (const float*)d_in[15];
  const float* bz      = (const float*)d_in[16];
  const float* wo      = (const float*)d_in[17];
  const float* bo      = (const float*)d_in[18];
  const float* wt      = (const float*)d_in[19];
  const float* bt      = (const float*)d_in[20];
  const float* lt2_w   = (const float*)d_in[21];
  const float* lt2_b   = (const float*)d_in[22];
  float* out = (float*)d_out;

  // workspace carve-up (float units, 256B-aligned chunks)
  float* base = (float*)d_ws;
  size_t off = 0;
  auto alloc = [&](size_t n) { float* p = base + off; off += (n + 63) & ~(size_t)63; return p; };
  const size_t BLH = (size_t)Bq * Lq * Hq;       // 3,276,800
  float*     invn      = alloc(NITEMS + 1);
  float*     hs        = alloc(Bq * Hq);
  float*     hsg       = alloc(Bq * Hq);
  float*     ht0       = alloc(Bq * Hq);
  float*     lasts     = alloc(Bq * 3 * Hq);
  float*     posproj   = alloc(Lq * Hq);
  _Float16*  hidden_h  = (_Float16*)alloc(BLH / 2);
  _Float16*  w1t_h     = (_Float16*)alloc(Hq * Hq / 2);
  _Float16*  glu1t_h   = (_Float16*)alloc(Hq * Hq / 2);
  _Float16*  wot_h     = (_Float16*)alloc(Hq * Hq / 2);
  _Float16*  wtt_h     = (_Float16*)alloc(Hq * Hq / 2);
  _Float16*  nh_h      = (_Float16*)alloc(BLH / 2);
  float*     gbuf      = alloc(BLH);
  float*     proj_o    = alloc(BLH);
  float*     proj_t    = alloc(BLH);
  float*     hts       = alloc(Bq * 3 * Hq);
  float*     proj_z    = alloc(Bq * 3 * Hq);
  float*     selbuf    = alloc(Bq * Hq);
  float*     aisbuf    = alloc(Bq * Hq);
  _Float16*  sess_h    = (_Float16*)alloc(Bq * Hq / 2);
  (void)ws_size; (void)in_sizes; (void)n_in; (void)out_size;

  // 1) item inverse norms (warms L2 with the embedding table)
  invnorm_kernel<<<(NITEMS + 7) / 8, 256, 0, stream>>>(emb_w, invn);
  // 2) per-batch prep
  prep_kernel<<<Bq, 128, 0, stream>>>(hidden, mask, glu2_w, hs, hsg, ht0, lasts);
  // 3) position projection (batch-invariant part of the w1 matmul)
  posproj_kernel<<<Lq, 128, 0, stream>>>(pos_w, w1, posproj);
  // 4) f16 conversions / weight transposes
  f16_convert_kernel<<<(int)((BLH + 255) / 256), 256, 0, stream>>>(hidden, hidden_h, (int)BLH);
  w_transpose_f16_kernel<<<64, 256, 0, stream>>>(w1 + Hq * Hq, w1t_h);   // hidden half of w1
  w_transpose_f16_kernel<<<64, 256, 0, stream>>>(glu1_w, glu1t_h);
  w_transpose_f16_kernel<<<64, 256, 0, stream>>>(wo, wot_h);
  w_transpose_f16_kernel<<<64, 256, 0, stream>>>(wt, wtt_h);
  // 5) WMMA stage matmuls over 25600 rows
  const int MB = (Bq * Lq) / 128;  // 200 blocks
  mm128_kernel<<<MB, 256, 0, stream>>>(hidden_h, w1t_h, 0, posproj, nullptr, nh_h, nullptr);
  mm128_kernel<<<MB, 256, 0, stream>>>(nh_h, glu1t_h, 1, glu1_b, hsg, nullptr, gbuf);
  mm128_kernel<<<MB, 256, 0, stream>>>(hidden_h, wot_h, 2, bo, nullptr, nullptr, proj_o);
  mm128_kernel<<<MB, 256, 0, stream>>>(hidden_h, wtt_h, 2, bt, nullptr, nullptr, proj_t);
  // 6) hts chain
  hts_kernel<<<Bq * 3, 128, 0, stream>>>(lasts, lq_w0, lq_b0, lq_w1, lq_b1, lq_w2, lq_b2, hts);
  htsnorm_kernel<<<Bq, 128, 0, stream>>>(hts);
  projz_kernel<<<Bq * 3, 128, 0, stream>>>(hts, wz, bz, proj_z);
  // 7) beta/select, attention, session
  beta_select_kernel<<<Bq, 128, 0, stream>>>(gbuf, w2, mask, hidden, selbuf);
  alpha_kernel<<<Bq, 256, 0, stream>>>(proj_z, proj_o, proj_t, mask, aisbuf);
  session_kernel<<<Bq, 128, 0, stream>>>(aisbuf, ht0, selbuf, lt2_w, lt2_b, sess_h);
  // 8) final scoring GEMM (memory-bound: ~512MB NT stores + L2-resident emb)
  final_gemm_kernel<<<(NITEMS + 127) / 128, 256, 0, stream>>>(emb_w, sess_h, invn, out);
}
